// ps_component_2413771620522
// MI455X (gfx1250) — compile-verified, matrix operand order checked by reading, not run on hardware
//
#include <hip/hip_runtime.h>
#include <hip/hip_bf16.h>

typedef __attribute__((ext_vector_type(16))) __bf16 v16bf;
typedef __attribute__((ext_vector_type(8)))  float  v8f;

// ---------------------------------------------------------------------------
// Transpose + convert 3x3 smooth-conv weights:
//   src OIHW f32 [cout][cin][3][3]  ->  dst bf16 [cout][tap][cin], tap=ky*3+kx
// K dimension (tap*128+cin) becomes contiguous per cout => B fragments are
// contiguous 16B loads.
// ---------------------------------------------------------------------------
__global__ void convert_smooth_weights(const float* __restrict__ w,
                                       __hip_bfloat16* __restrict__ wb) {
    int idx = blockIdx.x * blockDim.x + threadIdx.x;   // [cout][tap][cin]
    if (idx >= 128 * 9 * 128) return;
    int cout = idx / (9 * 128);
    int rem  = idx - cout * (9 * 128);
    int tap  = rem >> 7;
    int cin  = rem & 127;
    wb[idx] = __float2bfloat16(w[(cout * 128 + cin) * 9 + tap]);
}

// ---------------------------------------------------------------------------
// Initialize output [64,240] with the column-pool biases (re-done every call
// so graph replay stays deterministic).  Conv kernels atomically add into it.
// ---------------------------------------------------------------------------
__global__ void init_out(float* __restrict__ out,
                         const float* b1, const float* b2,
                         const float* b3, const float* b4) {
    int i = blockIdx.x * blockDim.x + threadIdx.x;
    if (i >= 64 * 240) return;
    int x = i % 240;
    float v = (x < 128) ? b1[0] : (x < 192) ? b2[0] : (x < 224) ? b3[0] : b4[0];
    out[i] = v;
}

// ---------------------------------------------------------------------------
// Fused 3x3 conv (128->128, SAME, bf16 WMMA, f32 acc) + bias + ReLU
// + 1x1 tidy reduction + column pooling, h-looped with a 3-row rolling
// LDS buffer (each input element read from HBM exactly once).
//
// Block: 256 threads = 8 waves laid out as 2 (M 32-blocks) x 4 (N 32-blocks).
// Each wave: 2x2 register blocking -> 4 WMMAs per K-chunk, 36 chunks.
// Weight (B) fragments are re-fetched from L2 every h iteration (72 frags =
// 576 VGPRs cannot live in registers); an opaque per-iteration integer offset
// blocks hoisting while keeping the pointers in the GLOBAL address space
// (plain pointer rewrites through asm degrade the loads to flat_*, which
// couple LOADcnt with DScnt and serialize against the LDS A-stream).
// ---------------------------------------------------------------------------
__global__ __launch_bounds__(256)
void conv3x3_tidy_pl_kernel(const float* __restrict__ p,            // [B,128,H,W]
                            const __hip_bfloat16* __restrict__ wb,  // [128][9][128] bf16
                            const float* __restrict__ bs,           // [128] conv bias
                            const float* __restrict__ wt,           // [128] tidy weight
                            const float* __restrict__ btp,          // [1]  tidy bias
                            const float* __restrict__ wpl,          // [H]  pool weight
                            float* __restrict__ out,                // [B,240]
                            int H, int W, int hchunk, int colOff) {
    constexpr int XS = 66;    // staged x extent: 64 + 2 halo
    constexpr int CS = 136;   // padded cin stride (bank-conflict-free staging)
    __shared__ __align__(16) __hip_bfloat16 raw[3 * XS * CS];  // ~54 KB
    __shared__ float tpart[64];

    const int tid  = threadIdx.x;
    const int wave = tid >> 5;
    const int lane = tid & 31;
    const int mi   = wave & 1;              // which x 32-block
    const int nj   = wave >> 1;             // which cout 32-block
    const int x0   = blockIdx.x * 64;
    const int h0   = blockIdx.y * hchunk;
    const int b    = blockIdx.z;

    const float* pb = p + (size_t)b * 128 * H * W;

    // stage one input row hr (f32 -> bf16) into LDS slot s, layout [s][x][cin]
    auto stage_row = [&](int hr, int s) {
        for (int idx = tid; idx < XS * 128; idx += 256) {
            int cin = idx / XS;
            int xi  = idx - cin * XS;       // 0..65 -> x = x0+xi-1
            int xs  = x0 + xi - 1;
            float v = 0.0f;
            if (hr >= 0 && hr < H && xs >= 0 && xs < W)
                v = pb[((size_t)cin * H + hr) * W + xs];
            raw[(s * XS + xi) * CS + cin] = __float2bfloat16(v);
        }
    };

    // rotating slot registers: s0 -> input row h-1, s1 -> h, s2 -> h+1
    int s0 = (h0 + 2) % 3;
    int s1 = h0 % 3;
    int s2 = (h0 + 1) % 3;

    stage_row(h0 - 1, s0);
    stage_row(h0,     s1);

    // per-lane constants
    const int kg  = (lane >> 4) * 8;        // K sub-group base (16-bit A/B layout)
    const int n0  = nj * 32 + (lane & 15);  // lane's two cout columns
    const int n1  = n0 + 16;
    const int a0x = mi * 32 + (lane & 15);  // lane's staged-x base for A0
    const int mb  = (lane >> 4) * 8;        // C/D row base within 16x16 tile

    const __hip_bfloat16* brow0 = wb + (size_t)n0 * 1152;
    const __hip_bfloat16* brow1 = wb + (size_t)n1 * 1152;
    __builtin_prefetch(brow0, 0, 2);        // global_prefetch_b8 (L2-resident K rows)
    __builtin_prefetch(brow1, 0, 2);

    const float bs0 = bs[n0], bs1 = bs[n1];
    const float wt0 = wt[n0], wt1 = wt[n1];
    const float bt  = btp[0];

    float cacc = 0.0f;   // per-column pool accumulator (tid < 64)

    for (int h = h0; h < h0 + hchunk; ++h) {
        stage_row(h + 1, s2);                // fill retiring slot (zeros past edge)
        if (tid < 64) tpart[tid] = 0.0f;
        __syncthreads();

        // Opaque zero offset: keeps B loads in the GLOBAL address space while
        // preventing the compiler from hoisting the 72 h-invariant fragments
        // (which would need 576 VGPRs and spill to scratch).
        int hoff = 0;
        asm volatile("" : "+v"(hoff));
        const __hip_bfloat16* b0base = brow0 + hoff;
        const __hip_bfloat16* b1base = brow1 + hoff;

        const int srow[3] = { s0, s1, s2 };

        v8f acc00 = {}, acc01 = {}, acc10 = {}, acc11 = {};
#pragma unroll
        for (int c = 0; c < 36; ++c) {
            const int tap  = c >> 2;                 // 0..8
            const int ky   = tap / 3;
            const int kx   = tap - ky * 3;
            const int cinb = (c & 3) * 32 + kg;      // cin base inside this tap

            const __hip_bfloat16* a0p = &raw[(srow[ky] * XS + a0x + kx) * CS + cinb];
            const __hip_bfloat16* a1p = a0p + 16 * CS;
            union { uint4 u[2]; v16bf v; } A0, A1, B0, B1;
            A0.u[0] = *(const uint4*)(a0p);
            A0.u[1] = *(const uint4*)(a0p + 16);
            A1.u[0] = *(const uint4*)(a1p);
            A1.u[1] = *(const uint4*)(a1p + 16);

            const __hip_bfloat16* b0p = b0base + c * 32 + kg;
            const __hip_bfloat16* b1p = b1base + c * 32 + kg;
            B0.u[0] = *(const uint4*)(b0p);
            B0.u[1] = *(const uint4*)(b0p + 16);
            B1.u[0] = *(const uint4*)(b1p);
            B1.u[1] = *(const uint4*)(b1p + 16);

            acc00 = __builtin_amdgcn_wmma_f32_16x16x32_bf16(
                false, A0.v, false, B0.v, (short)0, acc00, false, false);
            acc01 = __builtin_amdgcn_wmma_f32_16x16x32_bf16(
                false, A0.v, false, B1.v, (short)0, acc01, false, false);
            acc10 = __builtin_amdgcn_wmma_f32_16x16x32_bf16(
                false, A1.v, false, B0.v, (short)0, acc10, false, false);
            acc11 = __builtin_amdgcn_wmma_f32_16x16x32_bf16(
                false, A1.v, false, B1.v, (short)0, acc11, false, false);
        }

        // bias + ReLU + tidy reduction (combine the lane's two couts, then
        // LDS-atomic reduce across the 128 couts into 64 column partials).
#pragma unroll
        for (int r = 0; r < 8; ++r) {
            float v0 = wt0 * fmaxf(acc00[r] + bs0, 0.0f)
                     + wt1 * fmaxf(acc01[r] + bs1, 0.0f);
            float v1 = wt0 * fmaxf(acc10[r] + bs0, 0.0f)
                     + wt1 * fmaxf(acc11[r] + bs1, 0.0f);
            atomicAdd(&tpart[mi * 32 + mb + r],      v0);   // ds_add_f32
            atomicAdd(&tpart[mi * 32 + 16 + mb + r], v1);
        }
        __syncthreads();

        if (tid < 64) cacc += wpl[h] * (tpart[tid] + bt);

        // rotate slots: h advances by one input row
        int tsl = s0; s0 = s1; s1 = s2; s2 = tsl;
    }

    if (tid < 64 && (x0 + tid) < W)
        atomicAdd(out + (size_t)b * 240 + colOff + x0 + tid, cacc);
}

// ---------------------------------------------------------------------------
// p5 path (no smooth conv): c4[b,x] = sum_h wpl[h]*(sum_c wt[c]*p5[b,c,h,x]+bt)
// One wave per (b,x); lanes stride the 512 channels.  Adds onto init'd bias.
// ---------------------------------------------------------------------------
__global__ __launch_bounds__(256)
void p5_tidy_pl_kernel(const float* __restrict__ p5,   // [64,512,8,16]
                       const float* __restrict__ wt,   // [512]
                       const float* __restrict__ btp,  // [1]
                       const float* __restrict__ wpl,  // [8]
                       float* __restrict__ out) {
    int gwave = (blockIdx.x * blockDim.x + threadIdx.x) >> 5;
    int lane  = threadIdx.x & 31;
    if (gwave >= 64 * 16) return;
    int b = gwave >> 4;
    int x = gwave & 15;

    float sum = 0.0f;
    for (int c = lane; c < 512; c += 32) {
        const float* base = p5 + (((size_t)b * 512 + c) * 8) * 16 + x;
        float s = 0.0f;
#pragma unroll
        for (int hh = 0; hh < 8; ++hh) s += wpl[hh] * base[hh * 16];
        sum += wt[c] * s;
    }
#pragma unroll
    for (int off = 16; off > 0; off >>= 1)
        sum += __shfl_down(sum, off, 32);

    if (lane == 0) {
        float wplsum = 0.0f;
#pragma unroll
        for (int hh = 0; hh < 8; ++hh) wplsum += wpl[hh];
        out[(size_t)b * 240 + 224 + x] += sum + btp[0] * wplsum;
    }
}

// ---------------------------------------------------------------------------
extern "C" void kernel_launch(void* const* d_in, const int* in_sizes, int n_in,
                              void* d_out, int out_size, void* d_ws, size_t ws_size,
                              hipStream_t stream) {
    const float* p2 = (const float*)d_in[0];   // [64,128,64,128]
    const float* p3 = (const float*)d_in[1];   // [64,128,32,64]
    const float* p4 = (const float*)d_in[2];   // [64,128,16,32]
    const float* p5 = (const float*)d_in[3];   // [64,512,8,16]

    const float* w_s1 = (const float*)d_in[4];  const float* b_s1 = (const float*)d_in[5];
    const float* w_s2 = (const float*)d_in[6];  const float* b_s2 = (const float*)d_in[7];
    const float* w_s3 = (const float*)d_in[8];  const float* b_s3 = (const float*)d_in[9];
    const float* w_t1 = (const float*)d_in[10]; const float* b_t1 = (const float*)d_in[11];
    const float* w_t2 = (const float*)d_in[12]; const float* b_t2 = (const float*)d_in[13];
    const float* w_t3 = (const float*)d_in[14]; const float* b_t3 = (const float*)d_in[15];
    const float* w_t4 = (const float*)d_in[16]; const float* b_t4 = (const float*)d_in[17];
    const float* w_pl1 = (const float*)d_in[18]; const float* b_pl1 = (const float*)d_in[19];
    const float* w_pl2 = (const float*)d_in[20]; const float* b_pl2 = (const float*)d_in[21];
    const float* w_pl3 = (const float*)d_in[22]; const float* b_pl3 = (const float*)d_in[23];
    const float* w_pl4 = (const float*)d_in[24]; const float* b_pl4 = (const float*)d_in[25];

    float* out = (float*)d_out;

    // bf16 transposed weights in workspace: 3 x 147456 bf16
    __hip_bfloat16* wb1 = (__hip_bfloat16*)d_ws;
    __hip_bfloat16* wb2 = wb1 + 147456;
    __hip_bfloat16* wb3 = wb2 + 147456;

    const int WN = 128 * 9 * 128;
    convert_smooth_weights<<<(WN + 255) / 256, 256, 0, stream>>>(w_s1, wb1);
    convert_smooth_weights<<<(WN + 255) / 256, 256, 0, stream>>>(w_s2, wb2);
    convert_smooth_weights<<<(WN + 255) / 256, 256, 0, stream>>>(w_s3, wb3);

    init_out<<<(64 * 240 + 255) / 256, 256, 0, stream>>>(out, b_pl1, b_pl2, b_pl3, b_pl4);

    // p2: H=64, W=128 -> x-tiles 2, 4 h-chunks of 16  => 512 WGs
    conv3x3_tidy_pl_kernel<<<dim3(2, 4, 64), 256, 0, stream>>>(
        p2, wb1, b_s1, w_t1, b_t1, w_pl1, out, 64, 128, 16, 0);
    // p3: H=32, W=64  -> x-tiles 1, 4 h-chunks of 8   => 256 WGs
    conv3x3_tidy_pl_kernel<<<dim3(1, 4, 64), 256, 0, stream>>>(
        p3, wb2, b_s2, w_t2, b_t2, w_pl2, out, 32, 64, 8, 128);
    // p4: H=16, W=32  -> x-tiles 1 (half-wasted), 4 h-chunks of 4 => 256 WGs
    conv3x3_tidy_pl_kernel<<<dim3(1, 4, 64), 256, 0, stream>>>(
        p4, wb3, b_s3, w_t3, b_t3, w_pl3, out, 16, 32, 4, 192);

    // p5 path: 1024 waves = 128 blocks of 256
    p5_tidy_pl_kernel<<<128, 256, 0, stream>>>(p5, w_t4, b_t4, w_pl4, out);
}
